// GNNDMoN_807453851816
// MI455X (gfx1250) — matrix operand order repaired and checked by reading, not compiled, not run on hardware
//
#include <hip/hip_runtime.h>
#include <hip/hip_bf16.h>

typedef __attribute__((ext_vector_type(16))) _Float16 v16h;
typedef __attribute__((ext_vector_type(8)))  float    v8f;
typedef __attribute__((ext_vector_type(4)))  int      v4i;

typedef __attribute__((address_space(1))) v4i gv4i;   // global int4
typedef __attribute__((address_space(3))) v4i lv4i;   // LDS int4

#if __has_builtin(__builtin_amdgcn_global_load_async_to_lds_b128) && \
    __has_builtin(__builtin_amdgcn_s_wait_asynccnt)
#define HAS_ASYNC_LDS 1
#else
#define HAS_ASYNC_LDS 0
#endif

#define HID 64
#define KCLS 16
#define NPER 1024
#define NCLS 10

// ---------------------------------------------------------------------------
// Degree / normalization kernels
// ---------------------------------------------------------------------------
__global__ void k_init_deg(float* deg, int n) {
    int i = blockIdx.x * blockDim.x + threadIdx.x;
    if (i < n) deg[i] = 1.0f;  // self-loop
}

__global__ void k_count_deg(const int* __restrict__ dst, float* deg, int e) {
    int i = blockIdx.x * blockDim.x + threadIdx.x;
    if (i < e) atomicAdd(&deg[dst[i]], 1.0f);
}

__global__ void k_rsqrt(float* deg, int n) {
    int i = blockIdx.x * blockDim.x + threadIdx.x;
    if (i < n) deg[i] = rsqrtf(deg[i]);
}

// ---------------------------------------------------------------------------
// WMMA GEMM: out[N,64] = in[N,K] @ W[64,K]^T   (K = 128 or 64)
// One wave computes a 16-row x 64-col tile. Block = 8 waves = 128 rows.
// ---------------------------------------------------------------------------
__global__ void k_gemm_lin(const float* __restrict__ in,
                           const float* __restrict__ W,
                           float* __restrict__ out, int K) {
    const int tid  = threadIdx.x;
    const int wave = tid >> 5;
    const int lane = tid & 31;
    const int m    = lane & 15;
    const int hi   = lane >> 4;
    const int row0 = (blockIdx.x * 8 + wave) * 16;

    v8f c[4] = {};

    const int ksteps = K >> 5;
    for (int kt = 0; kt < ksteps; ++kt) {
        // A fragment: 16x32, lane m holds row (row0+m); K pairs per ISA layout
        v16h a;
#pragma unroll
        for (int v = 0; v < 8; ++v) {
            int kb = kt * 32 + ((v & 4) ? 16 : 0) + hi * 8 + (v & 3) * 2;
            const float* p = in + (size_t)(row0 + m) * K + kb;
            a[2 * v]     = (_Float16)p[0];
            a[2 * v + 1] = (_Float16)p[1];
        }
#pragma unroll
        for (int t = 0; t < 4; ++t) {
            // B fragment: 32x16; lane's column n = t*16+m, B[k][n] = W[n][k]
            v16h b;
#pragma unroll
            for (int v = 0; v < 8; ++v) {
                int k = kt * 32 + ((v & 4) ? 16 : 0) + hi * 8 + (v & 3) * 2;
                const float* p = W + (size_t)(t * 16 + m) * K + k;
                b[2 * v]     = (_Float16)p[0];
                b[2 * v + 1] = (_Float16)p[1];
            }
            c[t] = __builtin_amdgcn_wmma_f32_16x16x32_f16(
                false, a, false, b, (short)0, c[t], false, false);
        }
    }
    // C layout: VGPR v -> M = v + hi*8, N = t*16 + m
#pragma unroll
    for (int t = 0; t < 4; ++t)
#pragma unroll
        for (int v = 0; v < 8; ++v)
            out[(size_t)(row0 + v + hi * 8) * HID + t * 16 + m] = c[t][v];
}

// ---------------------------------------------------------------------------
// Per-graph GCN aggregation, fully in LDS. One block per (graph, feat-half).
// LDS: acc[1024x32] (128KB) + hs[1024x32] staged via async global->LDS
//      (128KB) + dl[1024] dinv slice (4KB)  => 260KB <= 320KB/WGP.
// out[node] = relu(sum_{e: dst=node} h[src]*dinv[src]*dinv[dst]
//                  + h[node]*dinv[node]^2 + bias)
// ---------------------------------------------------------------------------
__global__ void k_aggregate(const float* __restrict__ hlin,
                            const int* __restrict__ src,
                            const int* __restrict__ dst,
                            const float* __restrict__ dinv,
                            const float* __restrict__ bias,
                            float* __restrict__ out, int E) {
    extern __shared__ float lds[];
    float* acc = lds;                 // 1024*32 accumulators
    float* hs  = lds + NPER * 32;     // 1024*32 staged h half-tile
    float* dl  = hs  + NPER * 32;     // 1024    dinv slice

    const int g    = blockIdx.x >> 1;
    const int half = blockIdx.x & 1;
    const int tid  = threadIdx.x;     // 0..1023
    const int lo   = g * NPER;
    const int hiB  = lo + NPER;

    // Stage this graph's h half-tile into LDS (async global->LDS DMA path)
    {
        const float* gp = hlin + (size_t)(lo + tid) * HID + half * 32;
        float* lp = hs + (size_t)tid * 32;
#if HAS_ASYNC_LDS
        float* gpm = const_cast<float*>(gp);
#pragma unroll
        for (int i = 0; i < 8; ++i)
            __builtin_amdgcn_global_load_async_to_lds_b128(
                (gv4i*)(gpm + i * 4), (lv4i*)(lp + i * 4), 0, 0);
#else
#pragma unroll
        for (int i = 0; i < 32; ++i) lp[i] = gp[i];
#endif
        dl[tid] = dinv[lo + tid];
    }
    for (int i = tid; i < NPER * 32; i += NPER) acc[i] = 0.0f;
#if HAS_ASYNC_LDS
    __builtin_amdgcn_s_wait_asynccnt(0);
#endif
    __syncthreads();

    // Scan the full edge list (8MB, L2-resident); keep edges of this graph.
    for (int e = tid; e < E; e += NPER) {
        int s = src[e];
        if (s >= lo && s < hiB) {            // src & dst share the graph block
            int d = dst[e];
            float coef = dl[s - lo] * dl[d - lo];
            const float* hp = hs + (size_t)(s - lo) * 32;   // ds_load
            float* ap = acc + (size_t)(d - lo) * 32;
#pragma unroll
            for (int f = 0; f < 32; ++f)
                atomicAdd(&ap[f], hp[f] * coef);            // ds_add_f32
        }
    }
    __syncthreads();

    // Finalize: self-loop term + bias + relu
    const float di = dl[tid];
    const float d2 = di * di;
    const float* hp = hs + (size_t)tid * 32;
    float* op = out + (size_t)(lo + tid) * HID + half * 32;
#pragma unroll
    for (int f = 0; f < 32; ++f) {
        float v = acc[tid * 32 + f] + hp[f] * d2 + bias[half * 32 + f];
        op[f] = fmaxf(v, 0.0f);
    }
}

// ---------------------------------------------------------------------------
// DMoN head per graph: softmax assignment, WMMA s^T @ X, selu, mean, classify
// Block = 256 threads (8 waves), one block per graph.
// LDS: s[1024*16] + partials[8*16*64] + outv[64]
// ---------------------------------------------------------------------------
__global__ void k_dmon(const float* __restrict__ Xb,
                       const float* __restrict__ Wp,
                       const float* __restrict__ bp,
                       const float* __restrict__ Wl,
                       const float* __restrict__ bl,
                       float* __restrict__ out) {
    extern __shared__ float lds[];
    float* s_lds = lds;                      // 1024*16
    float* part  = lds + NPER * KCLS;        // 8*1024
    float* outv  = part + 8 * KCLS * HID;    // 64

    const int g   = blockIdx.x;
    const int tid = threadIdx.x;             // 0..255
    const float* Xg = Xb + (size_t)g * NPER * HID;

    // Step 1: cluster assignment s = softmax(X @ Wp^T + bp)  (scalar, tiny)
    for (int nl = tid; nl < NPER; nl += 256) {
        const float* xr = Xg + (size_t)nl * HID;
        float t[KCLS];
#pragma unroll
        for (int k = 0; k < KCLS; ++k) {
            float a = bp[k];
            const float* wr = Wp + k * HID;
            for (int h = 0; h < HID; ++h) a += xr[h] * wr[h];
            t[k] = a;
        }
        float mx = t[0];
#pragma unroll
        for (int k = 1; k < KCLS; ++k) mx = fmaxf(mx, t[k]);
        float sum = 0.0f;
#pragma unroll
        for (int k = 0; k < KCLS; ++k) { t[k] = __expf(t[k] - mx); sum += t[k]; }
        float inv = 1.0f / sum;
#pragma unroll
        for (int k = 0; k < KCLS; ++k) s_lds[nl * KCLS + k] = t[k] * inv;
    }
    __syncthreads();

    // Step 2: Xp_partial = s^T @ X over this wave's 128-node slice (WMMA)
    const int wave = tid >> 5;
    const int lane = tid & 31;
    const int m    = lane & 15;
    const int hi   = lane >> 4;

    v8f c[4] = {};
    for (int ks = 0; ks < 4; ++ks) {
        const int nbase = wave * 128 + ks * 32;
        // A = s^T : A[m=cluster][k=node], read from s_lds
        v16h a;
#pragma unroll
        for (int v = 0; v < 8; ++v) {
            int kk = ((v & 4) ? 16 : 0) + hi * 8 + (v & 3) * 2;
            a[2 * v]     = (_Float16)s_lds[(nbase + kk) * KCLS + m];
            a[2 * v + 1] = (_Float16)s_lds[(nbase + kk + 1) * KCLS + m];
        }
#pragma unroll
        for (int t = 0; t < 4; ++t) {
            // B = X : B[k=node][n=hid], lane column n = t*16+m
            v16h b;
#pragma unroll
            for (int v = 0; v < 8; ++v) {
                int kk = ((v & 4) ? 16 : 0) + hi * 8 + (v & 3) * 2;
                const float* p = Xg + (size_t)(nbase + kk) * HID + t * 16 + m;
                b[2 * v]     = (_Float16)p[0];
                b[2 * v + 1] = (_Float16)p[HID];
            }
            c[t] = __builtin_amdgcn_wmma_f32_16x16x32_f16(
                false, a, false, b, (short)0, c[t], false, false);
        }
    }
#pragma unroll
    for (int t = 0; t < 4; ++t)
#pragma unroll
        for (int v = 0; v < 8; ++v)
            part[wave * (KCLS * HID) + (v + hi * 8) * HID + t * 16 + m] = c[t][v];
    __syncthreads();

    // Step 3: reduce wave partials, apply selu -> Xp in part[0..1023]
    const float SC = 1.0507009873554805f, AL = 1.6732632423543772f;
    for (int i = tid; i < KCLS * HID; i += 256) {
        float a = 0.0f;
#pragma unroll
        for (int w = 0; w < 8; ++w) a += part[w * (KCLS * HID) + i];
        part[i] = SC * (a > 0.0f ? a : AL * (__expf(a) - 1.0f));
    }
    __syncthreads();

    // Step 4: mean over clusters
    if (tid < HID) {
        float a = 0.0f;
#pragma unroll
        for (int k = 0; k < KCLS; ++k) a += part[k * HID + tid];
        outv[tid] = a * (1.0f / KCLS);
    }
    __syncthreads();

    // Step 5: classifier
    if (tid < NCLS) {
        float a = bl[tid];
        const float* wr = Wl + tid * HID;
#pragma unroll
        for (int h = 0; h < HID; ++h) a += outv[h] * wr[h];
        out[g * NCLS + tid] = a;
    }
}

// ---------------------------------------------------------------------------
extern "C" void kernel_launch(void* const* d_in, const int* in_sizes, int n_in,
                              void* d_out, int out_size, void* d_ws, size_t ws_size,
                              hipStream_t stream) {
    const float* x  = (const float*)d_in[0];
    const int*   ei = (const int*)d_in[1];
    // d_in[2] = batch (unused: graphs are contiguous 1024-node blocks)
    const float* W1 = (const float*)d_in[3];
    const float* b1 = (const float*)d_in[4];
    const float* W2 = (const float*)d_in[5];
    const float* b2 = (const float*)d_in[6];
    const float* Wp = (const float*)d_in[7];
    const float* bp = (const float*)d_in[8];
    const float* Wl = (const float*)d_in[9];
    const float* bl = (const float*)d_in[10];
    float* out = (float*)d_out;

    const int N = in_sizes[0] / 128;   // 65536
    const int E = in_sizes[1] / 2;     // 1048576
    const int B = N / NPER;            // 64
    const int* srcp = ei;
    const int* dstp = ei + E;

    float* bufA = (float*)d_ws;                 // [N,64] linear output
    float* bufB = bufA + (size_t)N * HID;       // [N,64] activated output
    float* dinv = bufB + (size_t)N * HID;       // [N]    deg -> rsqrt(deg)

    const size_t aggShm  = (size_t)(NPER * 32 * 2 + NPER) * sizeof(float);   // 260 KB
    const size_t dmonShm = (size_t)(NPER * KCLS + 8 * KCLS * HID + HID) * sizeof(float);
    (void)hipFuncSetAttribute(reinterpret_cast<const void*>(k_aggregate),
                              hipFuncAttributeMaxDynamicSharedMemorySize, (int)aggShm);
    (void)hipFuncSetAttribute(reinterpret_cast<const void*>(k_dmon),
                              hipFuncAttributeMaxDynamicSharedMemorySize, (int)dmonShm);

    // degrees -> dinv
    k_init_deg<<<(N + 255) / 256, 256, 0, stream>>>(dinv, N);
    k_count_deg<<<(E + 255) / 256, 256, 0, stream>>>(dstp, dinv, E);
    k_rsqrt<<<(N + 255) / 256, 256, 0, stream>>>(dinv, N);

    // conv1: lin (WMMA) then LDS aggregate + relu
    k_gemm_lin<<<N / 128, 256, 0, stream>>>(x, W1, bufA, 128);
    k_aggregate<<<2 * B, NPER, aggShm, stream>>>(bufA, srcp, dstp, dinv, b1, bufB, E);

    // conv2
    k_gemm_lin<<<N / 128, 256, 0, stream>>>(bufB, W2, bufA, HID);
    k_aggregate<<<2 * B, NPER, aggShm, stream>>>(bufA, srcp, dstp, dinv, b2, bufB, E);

    // DMoN head + classifier
    k_dmon<<<B, 256, dmonShm, stream>>>(bufB, Wp, bp, Wl, bl, out);
}
